// MoE_Model_50766513439292
// MI455X (gfx1250) — matrix operand order repaired and checked by reading, not compiled, not run on hardware
//
#include <hip/hip_runtime.h>

typedef __attribute__((ext_vector_type(16))) __bf16 v16bf;
typedef __attribute__((ext_vector_type(8)))  float  v8f;

typedef unsigned int __attribute__((ext_vector_type(4))) tdm_v4u;
typedef int          __attribute__((ext_vector_type(8))) tdm_v8i;
typedef int          __attribute__((ext_vector_type(4))) tdm_v4i;

#define B_ROWS 8192
#define DIN 1024
#define DH  2048
#define DOUT 1024
#define NE 8
#define TAU_INV 1.25f

#if defined(__gfx1250__) && __has_builtin(__builtin_amdgcn_tensor_load_to_lds)
#define USE_TDM 1
#else
#define USE_TDM 0
#endif

__device__ __forceinline__ unsigned short f2bf(float f) {
  unsigned u = __float_as_uint(f);
  u += 0x7FFFu + ((u >> 16) & 1u);            // round-to-nearest-even
  return (unsigned short)(u >> 16);
}

union FragB16 { v16bf v; uint4 q[2]; };

#if USE_TDM
// Issue one TDM 2D-tile load: tile (32 x 128 rows) of bf16 from a row-major
// [rowsTotal x K] tensor into contiguous LDS ([128][32] row-major).
// D# layout per CDNA5 ISA section 8.3/8.4.
__device__ __forceinline__ void tdm_load_tile(unsigned lds_off,
                                              const unsigned short* gptr,
                                              unsigned K_elems, unsigned rowsTotal) {
  unsigned long long ga = (unsigned long long)(size_t)gptr;
  tdm_v4u g0;
  g0[0] = 1u;                                      // count=1, is_restore=0
  g0[1] = lds_off;                                 // lds_addr (bytes)
  g0[2] = (unsigned)ga;                            // global_addr[31:0]
  g0[3] = (unsigned)(ga >> 32) | 0x80000000u;      // global_addr[56:32] | type=2
  tdm_v8i g1;
  g1[0] = 0x00010000;                              // wg_mask=0, data_size=1 (2 bytes)
  g1[1] = (int)((K_elems & 0xFFFFu) << 16);        // tensor_dim0[15:0]
  g1[2] = (int)((K_elems >> 16) | ((rowsTotal & 0xFFFFu) << 16)); // dim0 hi | dim1 lo
  g1[3] = (int)((rowsTotal >> 16) | (32u << 16));  // dim1 hi | tile_dim0=32
  g1[4] = 128;                                     // tile_dim1=128, tile_dim2=0
  g1[5] = (int)K_elems;                            // tensor_dim0_stride[31:0]
  g1[6] = 0;                                       // stride0 hi | stride1 lo
  g1[7] = 0;                                       // stride1 hi
  tdm_v4i z4 = {0, 0, 0, 0};
#if __clang_major__ >= 23
  tdm_v8i z8 = {0, 0, 0, 0, 0, 0, 0, 0};
  __builtin_amdgcn_tensor_load_to_lds(g0, g1, z4, z4, z8, 0);
#else
  __builtin_amdgcn_tensor_load_to_lds(g0, g1, z4, z4, 0);
#endif
}
#endif

// ---------------- elementwise f32 -> bf16 (4-wide) ----------------
__global__ void __launch_bounds__(256) k_conv4(const float* __restrict__ src,
                                               unsigned short* __restrict__ dst, int n4) {
  int i = blockIdx.x * 256 + threadIdx.x;
  if (i < n4) {
    float4 f = ((const float4*)src)[i];
    ushort4 o;
    o.x = f2bf(f.x); o.y = f2bf(f.y); o.z = f2bf(f.z); o.w = f2bf(f.w);
    ((ushort4*)dst)[i] = o;
  }
}

// ------- tiled transpose + convert: src[e][R][C] f32 -> dst[e][C][R] bf16 -------
__global__ void __launch_bounds__(256) k_tconv(const float* __restrict__ src,
                                               unsigned short* __restrict__ dst,
                                               int R, int C) {
  __shared__ float tile[32][33];
  int e = blockIdx.z;
  int c0 = blockIdx.x * 32, r0 = blockIdx.y * 32;
  int tx = threadIdx.x, ty = threadIdx.y;
  const float* s = src + (size_t)e * R * C;
  unsigned short* d = dst + (size_t)e * R * C;
#pragma unroll
  for (int j = 0; j < 32; j += 8)
    tile[ty + j][tx] = s[(size_t)(r0 + ty + j) * C + (c0 + tx)];
  __syncthreads();
#pragma unroll
  for (int j = 0; j < 32; j += 8)
    d[(size_t)(c0 + ty + j) * R + (r0 + tx)] = f2bf(tile[tx][ty + j]);
}

// ---------------- gating: logits, gumbel-softmax probs, router probs ----------------
__global__ void __launch_bounds__(256) k_gate(const float* __restrict__ x,
                                              const float* __restrict__ gw,
                                              const float* __restrict__ gb,
                                              const float* __restrict__ gum,
                                              float* __restrict__ probs,
                                              float* __restrict__ rprobs) {
  int lane = threadIdx.x & 31;
  int row = blockIdx.x * 8 + (threadIdx.x >> 5);
  float a[8] = {0.f,0.f,0.f,0.f,0.f,0.f,0.f,0.f};
  const float* xr = x + (size_t)row * DIN;
  for (int c = lane; c < DIN; c += 32) {
    float xv = xr[c];
    float4 g0 = ((const float4*)(gw + (size_t)c * 8))[0];
    float4 g1 = ((const float4*)(gw + (size_t)c * 8))[1];
    a[0] += xv * g0.x; a[1] += xv * g0.y; a[2] += xv * g0.z; a[3] += xv * g0.w;
    a[4] += xv * g1.x; a[5] += xv * g1.y; a[6] += xv * g1.z; a[7] += xv * g1.w;
  }
#pragma unroll
  for (int e = 0; e < 8; ++e)
#pragma unroll
    for (int off = 16; off >= 1; off >>= 1)
      a[e] += __shfl_xor(a[e], off, 32);
  if (lane == 0) {
    float z1[8], z2[8];
    float m1 = -1e30f, m2 = -1e30f;
#pragma unroll
    for (int e = 0; e < 8; ++e) {
      float lg = a[e] + gb[e];
      z1[e] = (lg + gum[(size_t)row * 8 + e]) * TAU_INV;
      z2[e] = lg * TAU_INV;
      m1 = fmaxf(m1, z1[e]); m2 = fmaxf(m2, z2[e]);
    }
    float s1 = 0.f, s2 = 0.f;
#pragma unroll
    for (int e = 0; e < 8; ++e) {
      z1[e] = __expf(z1[e] - m1); s1 += z1[e];
      z2[e] = __expf(z2[e] - m2); s2 += z2[e];
    }
    float i1 = 1.f / s1, i2 = 1.f / s2;
#pragma unroll
    for (int e = 0; e < 8; ++e) {
      probs [(size_t)row * 8 + e] = z1[e] * i1;
      rprobs[(size_t)row * 8 + e] = z2[e] * i2;
    }
  }
}

// ---------------- deterministic per-expert column means ----------------
__global__ void __launch_bounds__(256) k_colmean(const float* __restrict__ probs,
                                                 const float* __restrict__ rprobs,
                                                 float* __restrict__ sums) {
  __shared__ float red[256];
  int e = blockIdx.x;
  int which = blockIdx.y;    // 0: importance (router), 1: load (probs)
  const float* s = which ? probs : rprobs;
  float acc = 0.f;
  for (int i = threadIdx.x; i < B_ROWS; i += 256)
    acc += s[(size_t)i * 8 + e];
  red[threadIdx.x] = acc;
  __syncthreads();
  for (int st = 128; st > 0; st >>= 1) {
    if (threadIdx.x < st) red[threadIdx.x] += red[threadIdx.x + st];
    __syncthreads();
  }
  if (threadIdx.x == 0) sums[which * 8 + e] = red[0] * (1.f / (float)B_ROWS);
}

__global__ void k_aux(const float* __restrict__ sums, float* __restrict__ out_aux) {
  if (threadIdx.x == 0) {
    const float* imp = sums;
    const float* ld  = sums + 8;
    float mi = 0.f, ml = 0.f;
#pragma unroll
    for (int e = 0; e < 8; ++e) { mi += imp[e]; ml += ld[e]; }
    mi *= 0.125f; ml *= 0.125f;
    float vi = 0.f, vl = 0.f, sw = 0.f;
#pragma unroll
    for (int e = 0; e < 8; ++e) {
      float di = imp[e] - mi, dl = ld[e] - ml;
      vi += di * di; vl += dl * dl;
      sw += imp[e] * ld[e];
    }
    float icv = sqrtf(vi / 7.f) / (mi + 1e-8f);
    float lcv = sqrtf(vl / 7.f) / (ml + 1e-8f);
    *out_aux = (8.f * sw + icv + lcv) * 0.05f;
  }
}

// ================= shared GEMM core =================
// Workgroup tile 128(M) x 128(N), BK=32, 8 waves: 4 in M x 2 in N,
// each wave owns a 32x64 tile = 2x4 accumulators of 16x16.
struct GemmIdx {
  int lane, wave, waveM, waveN, r, hi, colChunk, rowT;
};
__device__ __forceinline__ GemmIdx make_idx() {
  GemmIdx g;
  int tid = threadIdx.x;
  g.lane = tid & 31; g.wave = tid >> 5;
  g.waveM = g.wave & 3; g.waveN = g.wave >> 2;
  g.r = g.lane & 15; g.hi = g.lane >> 4;
  g.colChunk = (tid & 3) * 8; g.rowT = tid >> 2;
  return g;
}

#define TILE_HALFS (128 * 32)

__device__ __forceinline__ void gemm_core(const unsigned short* __restrict__ A,
                                          const unsigned short* __restrict__ Bt,
                                          int K, int rowsB,
                                          int mBase, int nBase,
                                          unsigned short* As, unsigned short* Bs,
                                          const GemmIdx& g, v8f acc[2][4]) {
  const int nSteps = K / 32;

  auto compute_step = [&](const unsigned short* Ab, const unsigned short* Bb) {
    FragB16 fa[2], fb[4];
#pragma unroll
    for (int im = 0; im < 2; ++im) {
      const uint4* pa = (const uint4*)(Ab + (g.waveM * 32 + im * 16 + g.r) * 32);
      fa[im].q[0] = pa[g.hi];     fa[im].q[1] = pa[g.hi + 2];
    }
#pragma unroll
    for (int j = 0; j < 4; ++j) {
      const uint4* pb = (const uint4*)(Bb + (g.waveN * 64 + j * 16 + g.r) * 32);
      fb[j].q[0] = pb[g.hi];      fb[j].q[1] = pb[g.hi + 2];
    }
#pragma unroll
    for (int im = 0; im < 2; ++im)
#pragma unroll
      for (int j = 0; j < 4; ++j)
        acc[im][j] = __builtin_amdgcn_wmma_f32_16x16x32_bf16(
            false, fa[im].v, false, fb[j].v, (short)0, acc[im][j], false, false);
  };

#if USE_TDM
  const unsigned ldsA = (unsigned)(size_t)As;
  const unsigned ldsB = (unsigned)(size_t)Bs;
  if (g.wave == 0) {
    tdm_load_tile(ldsA, A + (size_t)mBase * K, (unsigned)K, (unsigned)B_ROWS);
    tdm_load_tile(ldsB, Bt + (size_t)nBase * K, (unsigned)K, (unsigned)rowsB);
    if (nSteps > 1) {
      tdm_load_tile(ldsA + TILE_HALFS * 2, A + (size_t)mBase * K + 32, (unsigned)K, (unsigned)B_ROWS);
      tdm_load_tile(ldsB + TILE_HALFS * 2, Bt + (size_t)nBase * K + 32, (unsigned)K, (unsigned)rowsB);
    }
  }
  for (int i = 0; i < nSteps; ++i) {
    const int cur = i & 1;
    if (g.wave == 0) {
      if (i + 1 < nSteps) __builtin_amdgcn_s_wait_tensorcnt(2);
      else                __builtin_amdgcn_s_wait_tensorcnt(0);
    }
    __syncthreads();                 // buffer `cur` ready for everyone
    compute_step(As + cur * TILE_HALFS, Bs + cur * TILE_HALFS);
    __syncthreads();                 // everyone done reading buffer `cur`
    if (g.wave == 0 && i + 2 < nSteps) {
      int kt = (i + 2) * 32;
      tdm_load_tile(ldsA + cur * TILE_HALFS * 2, A + (size_t)mBase * K + kt, (unsigned)K, (unsigned)B_ROWS);
      tdm_load_tile(ldsB + cur * TILE_HALFS * 2, Bt + (size_t)nBase * K + kt, (unsigned)K, (unsigned)rowsB);
    }
  }
#else
  for (int i = 0; i < nSteps; ++i) {
    const int kt = i * 32;
    __syncthreads();
    {
      const uint4 va0 = *(const uint4*)(A + (size_t)(mBase + g.rowT) * K + kt + g.colChunk);
      const uint4 va1 = *(const uint4*)(A + (size_t)(mBase + g.rowT + 64) * K + kt + g.colChunk);
      const uint4 vb0 = *(const uint4*)(Bt + (size_t)(nBase + g.rowT) * K + kt + g.colChunk);
      const uint4 vb1 = *(const uint4*)(Bt + (size_t)(nBase + g.rowT + 64) * K + kt + g.colChunk);
      *(uint4*)(As + g.rowT * 32 + g.colChunk) = va0;
      *(uint4*)(As + (g.rowT + 64) * 32 + g.colChunk) = va1;
      *(uint4*)(Bs + g.rowT * 32 + g.colChunk) = vb0;
      *(uint4*)(Bs + (g.rowT + 64) * 32 + g.colChunk) = vb1;
      if (kt + 32 < K) {
        __builtin_prefetch(A + (size_t)(mBase + g.rowT) * K + kt + 32 + g.colChunk, 0, 0);
        __builtin_prefetch(Bt + (size_t)(nBase + g.rowT) * K + kt + 32 + g.colChunk, 0, 0);
      }
    }
    __syncthreads();
    compute_step(As, Bs);
  }
#endif
}

// ---------------- GEMM1: H = relu(Xbf @ W1T^T + b1), bf16 out ----------------
__global__ void __launch_bounds__(256) k_gemm1_relu(const unsigned short* __restrict__ A,
                                                    const unsigned short* __restrict__ Bt,
                                                    const float* __restrict__ bias,
                                                    unsigned short* __restrict__ H) {
  __shared__ unsigned short As[2 * TILE_HALFS];
  __shared__ unsigned short Bs[2 * TILE_HALFS];
  const GemmIdx g = make_idx();
  const int mBase = blockIdx.y * 128, nBase = blockIdx.x * 128;
  v8f acc[2][4] = {};
  gemm_core(A, Bt, DIN, DH, mBase, nBase, As, Bs, g, acc);

  const int mw = mBase + g.waveM * 32 + g.hi * 8;
#pragma unroll
  for (int j = 0; j < 4; ++j) {
    const int nw = nBase + g.waveN * 64 + j * 16 + g.r;
    const float bj = bias[nw];
#pragma unroll
    for (int v = 0; v < 8; ++v) {
      H[(size_t)(mw + v) * DH + nw]      = f2bf(fmaxf(acc[0][j][v] + bj, 0.f));
      H[(size_t)(mw + 16 + v) * DH + nw] = f2bf(fmaxf(acc[1][j][v] + bj, 0.f));
    }
  }
}

// ---------------- GEMM2: out (+)= (H @ W2T^T + b2) * probs[:,e] ----------------
__global__ void __launch_bounds__(256) k_gemm2_combine(const unsigned short* __restrict__ A,
                                                       const unsigned short* __restrict__ Bt,
                                                       const float* __restrict__ bias,
                                                       const float* __restrict__ probs,
                                                       int e, int init,
                                                       float* __restrict__ out) {
  __shared__ unsigned short As[2 * TILE_HALFS];
  __shared__ unsigned short Bs[2 * TILE_HALFS];
  __shared__ float pS[128];
  const GemmIdx g = make_idx();
  const int mBase = blockIdx.y * 128, nBase = blockIdx.x * 128;
  if (threadIdx.x < 128) pS[threadIdx.x] = probs[(size_t)(mBase + threadIdx.x) * NE + e];
  __syncthreads();
  v8f acc[2][4] = {};
  gemm_core(A, Bt, DH, DOUT, mBase, nBase, As, Bs, g, acc);

  const int mlw = g.waveM * 32 + g.hi * 8;
#pragma unroll
  for (int j = 0; j < 4; ++j) {
    const int nw = nBase + g.waveN * 64 + j * 16 + g.r;
    const float bj = bias[nw];
#pragma unroll
    for (int v = 0; v < 8; ++v) {
      const int ml0 = mlw + v, ml1 = mlw + 16 + v;
      const size_t o0 = (size_t)(mBase + ml0) * DOUT + nw;
      const size_t o1 = (size_t)(mBase + ml1) * DOUT + nw;
      const float v0 = (acc[0][j][v] + bj) * pS[ml0];
      const float v1 = (acc[1][j][v] + bj) * pS[ml1];
      if (init) { out[o0] = v0; out[o1] = v1; }
      else      { out[o0] += v0; out[o1] += v1; }
    }
  }
}

extern "C" void kernel_launch(void* const* d_in, const int* in_sizes, int n_in,
                              void* d_out, int out_size, void* d_ws, size_t ws_size,
                              hipStream_t stream) {
  const float* x   = (const float*)d_in[0];
  const float* gum = (const float*)d_in[1];
  const float* gw  = (const float*)d_in[2];
  const float* gb  = (const float*)d_in[3];
  const float* w1  = (const float*)d_in[4];
  const float* b1  = (const float*)d_in[5];
  const float* w2  = (const float*)d_in[6];
  const float* b2  = (const float*)d_in[7];
  float* out = (float*)d_out;

  char* ws = (char*)d_ws;
  unsigned short* xbf = (unsigned short*)(ws);                    // 16 MB
  unsigned short* w1T = (unsigned short*)(ws + 16777216ull);      // 32 MB  [E][Dh][Din]
  unsigned short* w2T = (unsigned short*)(ws + 50331648ull);      // 32 MB  [E][Dout][Dh]
  unsigned short* hbf = (unsigned short*)(ws + 83886080ull);      // 32 MB  [B][Dh]
  float* probs  = (float*)(ws + 117440512ull);                    // 256 KB
  float* rprobs = (float*)(ws + 117702656ull);                    // 256 KB
  float* sums   = (float*)(ws + 117964800ull);                    // 64 B

  // 1) precision conversion
  k_conv4<<<(B_ROWS * DIN / 4 + 255) / 256, 256, 0, stream>>>(x, xbf, B_ROWS * DIN / 4);
  k_tconv<<<dim3(DH / 32, DIN / 32, NE), dim3(32, 8), 0, stream>>>(w1, w1T, DIN, DH);
  k_tconv<<<dim3(DH / 32, DOUT / 32, NE), dim3(32, 8), 0, stream>>>(w2, w2T, DH, DOUT);

  // 2) gating + aux loss
  k_gate<<<B_ROWS / 8, 256, 0, stream>>>(x, gw, gb, gum, probs, rprobs);
  k_colmean<<<dim3(NE, 2), 256, 0, stream>>>(probs, rprobs, sums);
  k_aux<<<1, 32, 0, stream>>>(sums, out + (size_t)B_ROWS * DOUT);

  // 3) dense expert stack, fused combine
  for (int e = 0; e < NE; ++e) {
    k_gemm1_relu<<<dim3(DH / 128, B_ROWS / 128), 256, 0, stream>>>(
        xbf, w1T + (size_t)e * DH * DIN, b1 + (size_t)e * DH, hbf);
    k_gemm2_combine<<<dim3(DOUT / 128, B_ROWS / 128), 256, 0, stream>>>(
        hbf, w2T + (size_t)e * DOUT * DH, b2 + (size_t)e * DOUT, probs, e, e == 0 ? 1 : 0, out);
  }
}
